// _STFT_Internal_22754736734528
// MI455X (gfx1250) — matrix operand order repaired and checked by reading, not compiled, not run on hardware
//
#include <hip/hip_runtime.h>
#include <hip/hip_bf16.h>
#include <math.h>

// ---------------------------------------------------------------------------
// STFT power spectrogram as a WMMA GEMM on gfx1250 (wave32).
//   x:[32,1,480000] f32, hop=256, n_fft=1024, n_freq=513, frames=1872
//   out[b,freq,frame] = (sum_k win[k]cos(w_f k) x)^2 + (sum_k -win[k]sin(w_f k) x)^2
// ~126 GFLOP vs ~185MB HBM traffic (AI ~ 680 flop/B) -> compute bound on the
// matrix pipes: V_WMMA_F32_16X16X32_F16 with f32 accumulation over K=1024.
// ---------------------------------------------------------------------------

typedef __attribute__((ext_vector_type(8)))  _Float16 v8h;
typedef __attribute__((ext_vector_type(16))) _Float16 v16h;
typedef __attribute__((ext_vector_type(8)))  float    v8f;

#define NFFT      1024
#define NFREQ     513
#define FT_TILES  33              // ceil(513/16)
#define BATCH     32
#define SAMPLES   480000
#define HOP       256
#define NFRAMES   1872            // (480000-1024)/256 + 1
#define MT_TILES  117             // 1872/16
#define KSTEPS    32              // 1024/32
#define FRAG_ELEMS (FT_TILES * KSTEPS * 32 * 16)   // 540672 halves per matrix
#define OUT_BSTR  (NFREQ * NFRAMES)
#define LDSB_STR  24              // halves per k-row (48B: 16B-aligned, hits 2-cycle bank floor)

// ---------------------------------------------------------------------------
// Kernel 1: windowed DFT matrices (real & imag) in f16, pre-swizzled into the
// exact 16-bit A-matrix WMMA fragment layout:
//   ws[mat][ft][s][lane][j], freq = ft*16 + (lane&15), khalf = lane>>4,
//   k = s*32 + (j>>3)*16 + khalf*8 + (j&7).
// Angle reduced exactly: w_f*k = 2*pi*((freq*k) mod 1024)/1024.
// ---------------------------------------------------------------------------
__global__ void stft_dft_frag_kernel(const float* __restrict__ window,
                                     _Float16* __restrict__ dftR,
                                     _Float16* __restrict__ dftI) {
  int idx  = blockIdx.x * blockDim.x + threadIdx.x;     // 0 .. FRAG_ELEMS-1
  int j    = idx & 15;
  int lane = (idx >> 4) & 31;
  int s    = (idx >> 9) & 31;
  int ft   = idx >> 14;                                 // 33 * 2^14 == FRAG_ELEMS

  int freq  = ft * 16 + (lane & 15);
  int khalf = lane >> 4;
  int k     = s * 32 + ((j >> 3) << 4) + khalf * 8 + (j & 7);

  float r = 0.0f, im = 0.0f;
  if (freq < NFREQ) {
    int   m   = (freq * k) & (NFFT - 1);                // exact range reduction
    float ang = (float)m * (6.2831853071795864769f / (float)NFFT);
    float sn, cs;
    __sincosf(ang, &sn, &cs);
    float win = window[k];
    r  =  win * cs;
    im = -win * sn;
  }
  dftR[idx] = (_Float16)r;
  dftI[idx] = (_Float16)im;
}

// ---------------------------------------------------------------------------
// Kernel 2: main GEMM. One WG (256 thr = 8 waves) = (batch b, frame-tile nt).
// Stage 16 frames x 1024 samples into LDS (f16, k-major, padded rows) with
// b128 row stores, then each wave sweeps freq-tiles ft = wave, wave+8, ...
// accumulating a 16x16 f32 tile per DFT matrix over 32 WMMA k-steps.
// ---------------------------------------------------------------------------
__global__ __launch_bounds__(256)
void stft_wmma_kernel(const float* __restrict__ x,
                      const _Float16* __restrict__ dftR,
                      const _Float16* __restrict__ dftI,
                      const int* __restrict__ power_flag,
                      float* __restrict__ out) {
  const int b  = blockIdx.x / MT_TILES;
  const int nt = blockIdx.x % MT_TILES;

  // B tile in LDS: lB[k*24 + n], n = frame-in-tile (0..15). 48KB.
  __shared__ __align__(16) _Float16 lB[NFFT * LDSB_STR];

  const int tid  = threadIdx.x;
  const int lane = tid & 31;
  const int wave = tid >> 5;

  // Pull this wave's first A fragments toward L2/L0 while staging runs.
  {
    const size_t fo = ((size_t)(wave * KSTEPS) * 32 + lane) * 16;
    __builtin_prefetch(dftR + fo, 0, 3);
    __builtin_prefetch(dftI + fo, 0, 3);
  }

  // ---- Stage B: thread t owns k-rows {t, t+256, t+512, t+768}. For each row,
  // 16 coalesced f32 loads (lanes -> consecutive k), pack to f16, 2x ds b128.
  {
    const float* xb = x + (size_t)b * SAMPLES + (size_t)nt * 16 * HOP;
    #pragma unroll
    for (int r = 0; r < 4; ++r) {
      const int k = tid + r * 256;
      v8h lo, hi;
      #pragma unroll
      for (int f = 0; f < 8; ++f)  lo[f] = (_Float16)xb[f * HOP + k];
      #pragma unroll
      for (int f = 0; f < 8; ++f)  hi[f] = (_Float16)xb[(f + 8) * HOP + k];
      *(v8h*)&lB[k * LDSB_STR]     = lo;
      *(v8h*)&lB[k * LDSB_STR + 8] = hi;
    }
  }
  __syncthreads();

  const int pw = power_flag[0];

  for (int ft = wave; ft < FT_TILES; ft += 8) {
    v8f cr = {};
    v8f ci = {};
    const _Float16* pR = dftR + ((size_t)(ft * KSTEPS) * 32 + lane) * 16;
    const _Float16* pI = dftI + ((size_t)(ft * KSTEPS) * 32 + lane) * 16;

    #pragma unroll 4
    for (int s = 0; s < KSTEPS; ++s) {
      // A fragments: fully coalesced 2x b128 per matrix (pre-swizzled layout).
      v8h rlo = *(const v8h*)(pR);
      v8h rhi = *(const v8h*)(pR + 8);
      v8h ilo = *(const v8h*)(pI);
      v8h ihi = *(const v8h*)(pI + 8);
      // B fragment: lane holds n = lane&15, k = s*32 + 8*(lane>>4) + {0..7, 16..23} (table order).
      const _Float16* pB = &lB[(s * 32 + (lane >> 4) * 8) * LDSB_STR + (lane & 15)];
      v8h blo, bhi;
      #pragma unroll
      for (int e = 0; e < 8; ++e) { blo[e] = pB[e * LDSB_STR]; bhi[e] = pB[(16 + e) * LDSB_STR]; }

      v16h ar = __builtin_shufflevector(rlo, rhi, 0,1,2,3,4,5,6,7,8,9,10,11,12,13,14,15);
      v16h ai = __builtin_shufflevector(ilo, ihi, 0,1,2,3,4,5,6,7,8,9,10,11,12,13,14,15);
      v16h bf = __builtin_shufflevector(blo, bhi, 0,1,2,3,4,5,6,7,8,9,10,11,12,13,14,15);

      // D = A x B + C   (neg_a, A, neg_b, B, c_mod, C, reuse_a, reuse_b)
      cr = __builtin_amdgcn_wmma_f32_16x16x32_f16(false, ar, false, bf, (short)0, cr, false, false);
      ci = __builtin_amdgcn_wmma_f32_16x16x32_f16(false, ai, false, bf, (short)0, ci, false, false);

      pR += 32 * 16;
      pI += 32 * 16;
    }

    // C/D layout: VGPR r -> lanes 0-15: (M=r, N=lane), lanes 16-31: (M=r+8, N=lane-16).
    const int frow_hi = (lane >> 4) << 3;
    const int ncol    = lane & 15;
    float* ob = out + (size_t)b * OUT_BSTR + (size_t)(nt * 16 + ncol);
    #pragma unroll
    for (int r = 0; r < 8; ++r) {
      float re = cr[r], im = ci[r];
      float sv = re * re + im * im;
      if (!pw) sv = __builtin_amdgcn_sqrtf(sv);   // single v_sqrt_f32
      int freq = ft * 16 + r + frow_hi;
      if (freq < NFREQ) {
        // Write-once stream: non-temporal so x + DFT stay L2-resident.
        __builtin_nontemporal_store(sv, &ob[(size_t)freq * NFRAMES]);
      }
    }
  }
}

// ---------------------------------------------------------------------------
// Launch. Inputs: [0]=x f32, [1]=n, [2]=w, [3]=window f32, [4]=hop, [5]=power.
// d_ws holds the two pre-swizzled f16 DFT matrices (2 * 540672 halves ~ 2.1MB).
// ---------------------------------------------------------------------------
extern "C" void kernel_launch(void* const* d_in, const int* in_sizes, int n_in,
                              void* d_out, int out_size, void* d_ws, size_t ws_size,
                              hipStream_t stream) {
  const float* x      = (const float*)d_in[0];
  const float* window = (const float*)d_in[3];
  const int*   power  = (const int*)d_in[5];

  _Float16* dftR = (_Float16*)d_ws;
  _Float16* dftI = dftR + FRAG_ELEMS;

  // 1) Pre-swizzled windowed DFT fragments (f16) into scratch.
  stft_dft_frag_kernel<<<FRAG_ELEMS / 256, 256, 0, stream>>>(window, dftR, dftI);

  // 2) WMMA GEMM: 32 batches x 117 frame-tiles, 8 waves each over 33 freq-tiles.
  stft_wmma_kernel<<<BATCH * MT_TILES, 256, 0, stream>>>(
      x, dftR, dftI, power, (float*)d_out);
}
